// BEVencoder_87308095193873
// MI455X (gfx1250) — matrix operand order repaired and checked by reading, not compiled, not run on hardware
//
#include <hip/hip_runtime.h>

// ---------------- problem constants ----------------
#define BATCH   2
#define NCAMS   6
#define CDIM    256
#define IMH     32
#define IMW     88
#define NQ      10000
#define DP      4
#define HEADS   8
#define LEVELS  3
#define HD      32          // head dim == wave32 lanes
#define HW0     2816        // 32*88
#define HW1     704         // 16*44
#define HW2     176         // 8*22
#define HWT     3696
#define K32     8           // all GEMMs have K = 256 = 8 * 32

typedef __attribute__((ext_vector_type(16))) _Float16 v16h;
typedef __attribute__((ext_vector_type(8)))  float    v8f;

#define F_RELU 1
#define F_RES  2
#define F_HALF 4

// =====================================================================
// pack_b: convert f32 row-major B[K=256,N] into f16 WMMA B-fragments,
// fragment-contiguous: frag = nt*K32+kbi, 32 lanes x 16 halves each.
// One wave per fragment.
// =====================================================================
__global__ void __launch_bounds__(128)
pack_b(const float* __restrict__ Bw, int N, _Float16* __restrict__ out)
{
  const int nfrag = (N >> 4) * K32;
  const int frag = blockIdx.x * 4 + threadIdx.y;
  if (frag >= nfrag) return;
  const int nt = frag / K32, kbi = frag % K32;
  const int lane = threadIdx.x, half = lane >> 4, ml = lane & 15;
  const int col = nt * 16 + ml;
  v16h b;
#pragma unroll
  for (int r = 0; r < 8; ++r) {
    int k = kbi * 32 + ((r < 4) ? (8 * half + 2 * r) : (16 + 8 * half + 2 * (r - 4)));     // B frag: lanes0-15 K 0..7,16..23, lanes16-31 K 8..15,24..31
    b[2 * r]     = (_Float16)Bw[(long)k * N + col];
    b[2 * r + 1] = (_Float16)Bw[(long)(k + 1) * N + col];
  }
  *((v16h*)(out + (long)frag * 512 + lane * 16)) = b;
}

// =====================================================================
// WMMA GEMM, K=256, LDS-staged pre-packed B:
//   C[M,N] = A[M,256] * B[256,N] (+bias) (+residual) (relu) (f16 store)
// NT     : compile-time n-tiles per block; grid.x * NT == N/16 exactly,
//          so there is NO runtime tile predicate -> straight-line WMMA loop.
// AMODE 0: A row-major (lda), fragment = 4 x b128 loads per lane.
// AMODE 1: A channel-major (element (m,k) at A[m + k*sAk]), lane-coalesced.
// Block = 4 waves sharing one packed-B panel in LDS (NT*8*1KB).
// =====================================================================
template<int FLAGS, int AMODE, int NT>
__global__ void __launch_bounds__(128)
gemm_wmma_lds(const float* __restrict__ A, long lda, long sAk, long sAb,
              const _Float16* __restrict__ Bp,
              const float* __restrict__ bias, long sBiasB,
              const float* __restrict__ R,
              void* __restrict__ Cptr, int ldc, long sCb,
              int M)
{
  __shared__ _Float16 smem[NT * K32 * 512];
  const int lane = threadIdx.x, wy = threadIdx.y;
  const int nt0  = blockIdx.x * NT;

  // ---- cooperative copy of packed B panel to LDS (b128 both sides) ----
  {
    const uint4* src = (const uint4*)(Bp + (long)nt0 * K32 * 512);
    uint4* dst = (uint4*)smem;
#pragma unroll
    for (int i = 0; i < NT * K32 * 64 / 128; ++i)        // 64 x uint4 / frag
      dst[i * 128 + wy * 32 + lane] = src[i * 128 + wy * 32 + lane];
  }
  __syncthreads();

  const int mtile = blockIdx.y * 4 + wy;
  if (mtile * 16 >= M) return;                // wave-uniform
  const int z    = blockIdx.z;
  const int half = lane >> 4;
  const int ml   = lane & 15;
  const long mrow = (long)mtile * 16 + ml;
  const float* Ab    = A + (long)z * sAb;
  const float* biasb = bias ? (bias + (long)z * sBiasB) : nullptr;

  v8f acc[NT];
#pragma unroll
  for (int t = 0; t < NT; ++t) acc[t] = (v8f){0,0,0,0,0,0,0,0};

#pragma unroll
  for (int kbi = 0; kbi < K32; ++kbi) {
    // ---- A fragment (16x32 f16 layout, ISA 7.12.2) ----
    v16h a;
    if (AMODE == 0) {
      const float* ap = Ab + mrow * lda + kbi * 32 + 8 * half;
      float4 x0 = *(const float4*)ap;
      float4 x1 = *(const float4*)(ap + 4);
      float4 x2 = *(const float4*)(ap + 16);
      float4 x3 = *(const float4*)(ap + 20);
      a[0]=(_Float16)x0.x; a[1]=(_Float16)x0.y; a[2]=(_Float16)x0.z; a[3]=(_Float16)x0.w;
      a[4]=(_Float16)x1.x; a[5]=(_Float16)x1.y; a[6]=(_Float16)x1.z; a[7]=(_Float16)x1.w;
      a[8]=(_Float16)x2.x; a[9]=(_Float16)x2.y; a[10]=(_Float16)x2.z; a[11]=(_Float16)x2.w;
      a[12]=(_Float16)x3.x; a[13]=(_Float16)x3.y; a[14]=(_Float16)x3.z; a[15]=(_Float16)x3.w;
    } else {
#pragma unroll
      for (int r = 0; r < 8; ++r) {
        int k0 = kbi * 32 + ((r < 4) ? (8 * half + 2 * r) : (16 + 8 * half + 2 * (r - 4)));
        a[2 * r]     = (_Float16)Ab[mrow + (long)k0       * sAk];
        a[2 * r + 1] = (_Float16)Ab[mrow + (long)(k0 + 1) * sAk];
      }
    }
    // ---- NT WMMAs, B fragments from LDS (no predicates) ----
#pragma unroll
    for (int t = 0; t < NT; ++t) {
      v16h b = *((const v16h*)(smem + ((t * K32 + kbi) * 512) + lane * 16));
      acc[t] = __builtin_amdgcn_wmma_f32_16x16x32_f16(
          false, a, false, b, (short)0, acc[t], false, false);
    }
  }

  // ---- epilogue: C layout vgpr r -> row = r + 8*half, col = lane&15 ----
#pragma unroll
  for (int t = 0; t < NT; ++t) {
    int col = (nt0 + t) * 16 + ml;
    float bv = biasb ? biasb[col] : 0.f;
    v8f c = acc[t];
#pragma unroll
    for (int r = 0; r < 8; ++r) {
      long row = (long)mtile * 16 + r + 8 * half;
      float val = c[r] + bv;
      if (FLAGS & F_RES)  val += R[row * (long)ldc + col];
      if (FLAGS & F_RELU) val = fmaxf(val, 0.f);
      if (FLAGS & F_HALF)
        ((_Float16*)Cptr)[(long)z * sCb + row * (long)ldc + col] = (_Float16)val;
      else
        ((float*)Cptr)[(long)z * sCb + row * (long)ldc + col] = val;
    }
  }
}

// ---------------- 2x2 average pool (channel-major) ----------------
__global__ void avgpool2(const float* __restrict__ in, float* __restrict__ out,
                         int hin, int win, int total)
{
  int i = blockIdx.x * blockDim.x + threadIdx.x;
  if (i >= total) return;
  int wout = win >> 1, hout = hin >> 1;
  int x = i % wout; int r = i / wout; int y = r % hout; int ch = r / hout;
  const float* p = in + ((long)ch * hin + 2 * y) * win + 2 * x;
  out[i] = 0.25f * (p[0] + p[1] + p[win] + p[win + 1]);
}

// ---- combined bias: (b_fpn[l]+cam[n]+lvl[l]) @ w_value + b_value  (18x256) ----
__global__ void bias_combine(const float* __restrict__ bfpn, const float* __restrict__ cams,
                             const float* __restrict__ lvls, const float* __restrict__ wval,
                             const float* __restrict__ bval, float* __restrict__ outb)
{
  int nl = blockIdx.x;            // n*3 + l
  int n = nl / 3, l = nl % 3;
  int e = threadIdx.x;
  float s = bval[e];
  for (int d = 0; d < CDIM; ++d) {
    float t = bfpn[l * CDIM + d] + cams[n * CDIM + d] + lvls[l * CDIM + d];
    s += t * wval[(long)d * CDIM + e];
  }
  outb[(long)nl * CDIM + e] = s;
}

// ---------------- reference-point projection + validity ----------------
__global__ void project_ref(const float* __restrict__ l2i,
                            float* __restrict__ ref, float* __restrict__ valid)
{
  long i = (long)blockIdx.x * blockDim.x + threadIdx.x;   // over B*NCAMS*NQ
  if (i >= (long)BATCH * NCAMS * NQ) return;
  int q = (int)(i % NQ); long bn = i / NQ;
  int ix = q % 100, iy = q / 100;
  float gx = (ix + 0.5f) * 0.01f, gy = (iy + 0.5f) * 0.01f;
  float X = gx * 102.4f - 51.2f, Y = gy * 102.4f - 51.2f;
  const float* P = l2i + bn * 16;
  float p00=P[0],p01=P[1],p02=P[2],p03=P[3];
  float p10=P[4],p11=P[5],p12=P[6],p13=P[7];
  float p20=P[8],p21=P[9],p22=P[10],p23=P[11];
  int any = 0;
#pragma unroll
  for (int d = 0; d < DP; ++d) {
    float zn = (0.5f + d * (7.f / 3.f)) * 0.125f;   // linspace(0.5,7.5,4)/8
    float Z  = zn * 8.f - 5.f;
    float px = p00 * X + p01 * Y + p02 * Z + p03;
    float py = p10 * X + p11 * Y + p12 * Z + p13;
    float pz = p20 * X + p21 * Y + p22 * Z + p23;
    float dep = fmaxf(pz, 1e-5f);
    float rx = (px / dep) * (1.f / 704.f);
    float ry = (py / dep) * (1.f / 256.f);
    any |= (pz > 1e-5f) & (rx > 0.f) & (rx < 1.f) & (ry > 0.f) & (ry < 1.f);
    ref[(i * DP + d) * 2 + 0] = rx;
    ref[(i * DP + d) * 2 + 1] = ry;
  }
  valid[i] = any ? 1.f : 0.f;
}

// ---------------- q = query + pos ----------------
__global__ void addvec(const float* __restrict__ a, const float* __restrict__ b,
                       float* __restrict__ o, int n)
{
  int i = blockIdx.x * blockDim.x + threadIdx.x;
  if (i < n) o[i] = a[i] + b[i];
}

// ---------------- softmax over 12 (levels*pillars) per (q,head) ----------------
__global__ void softmax12(float* __restrict__ lg)
{
  int i = blockIdx.x * blockDim.x + threadIdx.x;   // NQ*HEADS
  if (i >= NQ * HEADS) return;
  float* p = lg + (long)i * 12;
  float v[12], mx = p[0];
#pragma unroll
  for (int j = 0; j < 12; ++j) { v[j] = p[j]; mx = fmaxf(mx, v[j]); }
  float s = 0.f;
#pragma unroll
  for (int j = 0; j < 12; ++j) { v[j] = __expf(v[j] - mx); s += v[j]; }
  float inv = 1.f / s;
#pragma unroll
  for (int j = 0; j < 12; ++j) p[j] = v[j] * inv;
}

// ---------------- deformable bilinear sampling + cam-masked average ----------------
// block = (32 lanes = head-dim, 8 waves = heads); one block per query.
// Each gather is 32 consecutive f16 (64B) -> coalesced per wave.
__global__ void __launch_bounds__(256)
deform_sample(const _Float16* __restrict__ vbuf, const float* __restrict__ ref,
              const float* __restrict__ valid, const float* __restrict__ offs,
              const float* __restrict__ attn, float* __restrict__ slots, int batch)
{
  const int q = blockIdx.x, h = threadIdx.y, lane = threadIdx.x;
  const int hwoff[LEVELS] = {0, HW0, HW0 + HW1};
  const int lw[LEVELS] = {88, 44, 22};
  const int lh[LEVELS] = {32, 16, 8};

  float aw[12], ox[12], oy[12];
#pragma unroll
  for (int j = 0; j < 12; ++j) {
    long base = ((long)q * HEADS + h) * 12 + j;
    aw[j] = attn[base];
    ox[j] = offs[base * 2 + 0];
    oy[j] = offs[base * 2 + 1];
  }

  float acc = 0.f; int cnt = 0;
  for (int n = 0; n < NCAMS; ++n) {
    long bn = (long)batch * NCAMS + n;
    if (valid[bn * NQ + q] == 0.f) continue;   // zero contribution anyway
    cnt++;
    const float* rp = ref + (bn * NQ + q) * (DP * 2);
#pragma unroll
    for (int l = 0; l < LEVELS; ++l) {
      const int wl = lw[l], hl = lh[l];
      const float iwl = 1.f / wl, ihl = 1.f / hl;
      const _Float16* vl = vbuf + ((bn * HWT + hwoff[l]) * (long)CDIM) + h * HD + lane;
#pragma unroll
      for (int d = 0; d < DP; ++d) {
        int j = l * DP + d;
        float x = (rp[d * 2 + 0] + ox[j] * iwl) * wl - 0.5f;
        float y = (rp[d * 2 + 1] + oy[j] * ihl) * hl - 0.5f;
        float x0 = floorf(x), y0 = floorf(y);
        float a = aw[j];
#pragma unroll
        for (int dy = 0; dy < 2; ++dy)
#pragma unroll
          for (int dx = 0; dx < 2; ++dx) {
            float xi = x0 + dx, yi = y0 + dy;
            if (xi >= 0.f && xi < (float)wl && yi >= 0.f && yi < (float)hl) {
              float wgt = (1.f - fabsf(x - xi)) * (1.f - fabsf(y - yi));
              long idx = (long)((int)yi * wl + (int)xi);
              acc += a * wgt * (float)vl[idx * CDIM];
            }
          }
      }
    }
  }
  slots[(long)q * CDIM + h * HD + lane] = acc / (float)(cnt > 0 ? cnt : 1);
}

// ---------------- final outputs: [ego | ego | nb] ----------------
__global__ void write_out(const float* __restrict__ ego, const float* __restrict__ nb,
                          float* __restrict__ out)
{
  int i = blockIdx.x * blockDim.x + threadIdx.x;
  if (i >= NQ * CDIM) return;
  float e = ego[i];
  out[i]                 = e;            // bq[None]
  out[NQ * CDIM + i]     = e;            // bev_groups[0] (raw reshape of ego)
  out[2 * NQ * CDIM + i] = nb[i];        // bev_groups[1]
}

// =====================================================================
extern "C" void kernel_launch(void* const* d_in, const int* in_sizes, int n_in,
                              void* d_out, int out_size, void* d_ws, size_t ws_size,
                              hipStream_t stream)
{
  const float* imgs   = (const float*)d_in[0];
  const float* l2i    = (const float*)d_in[1];
  const float* bquery = (const float*)d_in[2];
  const float* bpos   = (const float*)d_in[3];
  const float* cams   = (const float*)d_in[4];
  const float* lvls   = (const float*)d_in[5];
  const float* w_fpn  = (const float*)d_in[6];
  const float* b_fpn  = (const float*)d_in[7];
  const float* w_val  = (const float*)d_in[8];
  const float* b_val  = (const float*)d_in[9];
  const float* w_off  = (const float*)d_in[10];
  const float* b_off  = (const float*)d_in[11];
  const float* w_attn = (const float*)d_in[12];
  const float* b_attn = (const float*)d_in[13];
  const float* w_out  = (const float*)d_in[14];
  const float* b_out  = (const float*)d_in[15];
  const float* w_ts1  = (const float*)d_in[16];
  const float* b_ts1  = (const float*)d_in[17];
  const float* w_ts2  = (const float*)d_in[18];
  const float* b_ts2  = (const float*)d_in[19];
  float* out = (float*)d_out;

  // ---- workspace layout (float offsets) ----
  float*    ws   = (float*)d_ws;
  _Float16* wsh  = (_Float16*)d_ws;
  _Float16* vbuf = wsh;                             // 11,354,112 halves
  float* ref    = ws + 5677056;                     //   960,000
  float* validb = ws + 6637056;                     //   120,000
  float* bq1    = ws + 6757056;                     // 2,560,000
  float* bq2    = ws + 9317056;
  float* bq3    = ws + 11877056;
  float* qpos   = ws + 14437056;
  float* offb   = ws + 16997056;                    // 1,920,000
  float* attnb  = ws + 18917056;                    //   960,000
  float* slots  = ws + 19877056;
  float* tbuf   = ws + 22437056;
  float* ubuf   = ws + 24997056;                    // ends 27,557,056 floats
  // prep-phase scratch overlaps the round-phase transient region:
  float* pool1  = qpos;                             // 2,162,688
  float* pool2  = pool1 + 2162688;                  //   540,672
  float* wcomb  = pool2 + 540672;                   //   196,608 (3 x 256 x 256)
  float* biascb = wcomb + 196608;                   //     4,608 (18 x 256)
  // persistent packed-f16 weight fragments (after ubuf), 32B-aligned:
  _Float16* pk_wval  = wsh + 2L * 27557056;         //  65,536 halves
  _Float16* pk_woff  = pk_wval  + 65536;            //  49,152
  _Float16* pk_wattn = pk_woff  + 49152;            //  24,576
  _Float16* pk_wout  = pk_wattn + 24576;            //  65,536
  _Float16* pk_wts1  = pk_wout  + 65536;            //  65,536
  _Float16* pk_wts2  = pk_wts1  + 65536;            //  65,536
  _Float16* pk_wcomb = pk_wts2  + 65536;            // 196,608 (3 levels)

  const dim3 gblk(32, 4);

  // ---- 1) image pyramids ----
  avgpool2<<<(2162688 + 255) / 256, 256, 0, stream>>>(imgs, pool1, IMH, IMW, 2162688);
  avgpool2<<<(540672 + 255) / 256, 256, 0, stream>>>(pool1, pool2, 16, 44, 540672);

  // ---- 2) pack static weights into WMMA B-fragments (f16) ----
  pack_b<<<32, gblk, 0, stream>>>(w_val,  256, pk_wval);
  pack_b<<<24, gblk, 0, stream>>>(w_off,  192, pk_woff);
  pack_b<<<12, gblk, 0, stream>>>(w_attn,  96, pk_wattn);
  pack_b<<<32, gblk, 0, stream>>>(w_out,  256, pk_wout);
  pack_b<<<32, gblk, 0, stream>>>(w_ts1,  256, pk_wts1);
  pack_b<<<32, gblk, 0, stream>>>(w_ts2,  256, pk_wts2);

  // ---- 3) fold FPN into value projection: Wcomb[l] = w_fpn[l] @ w_value ----
  gemm_wmma_lds<0, 0, 4><<<dim3(4, 4, 3), gblk, 0, stream>>>(
      w_fpn, 256, 0, 65536, pk_wval, nullptr, 0, nullptr,
      (void*)wcomb, 256, 65536, 256);
  bias_combine<<<18, 256, 0, stream>>>(b_fpn, cams, lvls, w_val, b_val, biascb);
  for (int l = 0; l < LEVELS; ++l)
    pack_b<<<32, gblk, 0, stream>>>(wcomb + (long)l * 65536, 256,
                                    pk_wcomb + (long)l * 65536);

  // ---- 4) v[b,n,hw,c] = pooled(channel-major) @ Wcomb + biascb  -> f16 ----
  const int   hwl[LEVELS]   = {HW0, HW1, HW2};
  const int   hwoff[LEVELS] = {0, HW0, HW0 + HW1};
  const float* srcs[LEVELS] = {imgs, pool1, pool2};
  for (int b = 0; b < BATCH; ++b)
    for (int l = 0; l < LEVELS; ++l) {
      int hw = hwl[l];
      const float* Asrc = srcs[l] + (long)b * NCAMS * CDIM * hw;
      void* Cdst = (void*)(vbuf + ((long)b * NCAMS * HWT + hwoff[l]) * CDIM);
      dim3 grid(4, (hw / 16 + 3) / 4, NCAMS);
      gemm_wmma_lds<F_HALF, 1, 4><<<grid, gblk, 0, stream>>>(
          Asrc, 0, hw, (long)CDIM * hw,                 // A channel-major
          pk_wcomb + (long)l * 65536,
          biascb + l * 256, 768,                        // bias per (n,l)
          nullptr, Cdst, 256, (long)HWT * CDIM, hw);
    }

  // ---- 5) reference projection + per-(b,n,q) validity ----
  project_ref<<<(BATCH * NCAMS * NQ + 255) / 256, 256, 0, stream>>>(l2i, ref, validb);

  // ---- 6) three sca+tsblock rounds (b=0, b=0, b=1) ----
  const float* qins[3]  = {bquery, bq1, bq2};
  float*       qouts[3] = {bq1, bq2, bq3};
  const int    batches[3] = {0, 0, 1};
  for (int r = 0; r < 3; ++r) {
    const float* qin = qins[r];
    addvec<<<(NQ * CDIM + 255) / 256, 256, 0, stream>>>(qin, bpos, qpos, NQ * CDIM);
    // offsets: 10000x192  (3 blocks x NT=4 = 12 tiles)
    gemm_wmma_lds<0, 0, 4><<<dim3(3, 157, 1), gblk, 0, stream>>>(
        qpos, 256, 0, 0, pk_woff, b_off, 0, nullptr,
        (void*)offb, 192, 0, NQ);
    // attention logits: 10000x96  (2 blocks x NT=3 = 6 tiles)
    gemm_wmma_lds<0, 0, 3><<<dim3(2, 157, 1), gblk, 0, stream>>>(
        qpos, 256, 0, 0, pk_wattn, b_attn, 0, nullptr,
        (void*)attnb, 96, 0, NQ);
    softmax12<<<(NQ * HEADS + 255) / 256, 256, 0, stream>>>(attnb);
    // deformable sampling + masked cam average
    deform_sample<<<NQ, dim3(32, 8), 0, stream>>>(vbuf, ref, validb, offb, attnb,
                                                  slots, batches[r]);
    // out-proj + residual
    gemm_wmma_lds<F_RES, 0, 4><<<dim3(4, 157, 1), gblk, 0, stream>>>(
        slots, 256, 0, 0, pk_wout, b_out, 0, qin,
        (void*)tbuf, 256, 0, NQ);
    // tsblock: relu(x@w1+b1)@w2+b2
    gemm_wmma_lds<F_RELU, 0, 4><<<dim3(4, 157, 1), gblk, 0, stream>>>(
        tbuf, 256, 0, 0, pk_wts1, b_ts1, 0, nullptr,
        (void*)ubuf, 256, 0, NQ);
    gemm_wmma_lds<0, 0, 4><<<dim3(4, 157, 1), gblk, 0, stream>>>(
        ubuf, 256, 0, 0, pk_wts2, b_ts2, 0, nullptr,
        (void*)qouts[r], 256, 0, NQ);
  }

  // ---- 7) outputs: bq (=ego), bev_groups[0]=ego, bev_groups[1]=nb ----
  write_out<<<(NQ * CDIM + 255) / 256, 256, 0, stream>>>(bq2, bq3, out);
}
